// EGNN_md17_2774548873301
// MI455X (gfx1250) — matrix operand order repaired and checked by reading, not compiled, not run on hardware
//
#include <hip/hip_runtime.h>
#include <hip/hip_bf16.h>
#include <stdint.h>

#define HID   64
#define NPOSC 10
#define NCOVC 20

typedef __bf16 bf16_t;
typedef bf16_t v16bf __attribute__((ext_vector_type(16)));
typedef float  v8f   __attribute__((ext_vector_type(8)));

union B16 { v16bf v; unsigned u[8]; };

__device__ __forceinline__ unsigned bf16r(float f){
  unsigned u = __float_as_uint(f);
  return (u + 0x7fffu + ((u >> 16) & 1u)) >> 16;
}
__device__ __forceinline__ unsigned bf16pk(float lo, float hi){
  return bf16r(lo) | (bf16r(hi) << 16);
}
__device__ __forceinline__ unsigned short bf16s(float f){ return (unsigned short)bf16r(f); }

__device__ __forceinline__ void wave_sync(){
  asm volatile("" ::: "memory");
  __builtin_amdgcn_wave_barrier();
  asm volatile("" ::: "memory");
}

__device__ __forceinline__ v8f vzero8(){
  v8f z;
#pragma unroll
  for (int i = 0; i < 8; ++i) z[i] = 0.f;
  return z;
}

__device__ __forceinline__ v8f wmma_bf16(v16bf a, v16bf b, v8f c){
  // D = A(16x32 bf16) * B(32x16 bf16) + C(16x16 f32)
  return __builtin_amdgcn_wmma_f32_16x16x32_bf16(false, a, false, b, (short)0, c, false, false);
}

// Pre-swizzled B operand: [(t*nChunks+c)][lane][8 dwords] contiguous
__device__ __forceinline__ v16bf loadB(const unsigned* __restrict__ base, int idx, int lane){
  const uint4* p = (const uint4*)(base + (((size_t)idx * 32 + lane) << 3));
  uint4 a = p[0], b = p[1];
  B16 r;
  r.u[0]=a.x; r.u[1]=a.y; r.u[2]=a.z; r.u[3]=a.w;
  r.u[4]=b.x; r.u[5]=b.y; r.u[6]=b.z; r.u[7]=b.w;
  return r.v;
}

__device__ __forceinline__ void fillA_u(B16& A, uint4 q0, uint4 q1){
  A.u[0]=q0.x; A.u[1]=q0.y; A.u[2]=q0.z; A.u[3]=q0.w;
  A.u[4]=q1.x; A.u[5]=q1.y; A.u[6]=q1.z; A.u[7]=q1.w;
}

__device__ __forceinline__ float silu(float x){ return x / (1.f + __expf(-x)); }

// ---------------- weight pre-swizzle into WMMA B layout (bf16) ----------------
__global__ void prep_wB(const float* __restrict__ w, unsigned* __restrict__ dstB,
                        int K, int Nw, int nTiles, int nChunks){
  int gid = blockIdx.x * blockDim.x + threadIdx.x;
  int total = nTiles * nChunks * 256;
  if (gid >= total) return;
  int d    = gid & 7;
  int lane = (gid >> 3) & 31;
  int tc   = gid >> 8;              // == t*nChunks + c
  int half = lane >> 4;
  int t    = tc / nChunks;
  int n    = t*16 + (lane & 15);
  int k0   = 32 * (tc % nChunks) + 2*(d&3) + 16*(d>>2) + half*8;   // B layout: lane=N col, VGPR d = K pair (table order)
  float f0 = (k0   < K && n < Nw) ? w[(size_t)k0    * Nw + n] : 0.f;
  float f1 = (k0+1 < K && n < Nw) ? w[(size_t)(k0+1)* Nw + n] : 0.f;
  dstB[gid] = bf16pk(f0, f1);
}

// ---------------- embed: h = MLP2(|vel| per frame) ----------------
__global__ __launch_bounds__(64) void egnn_embed(
    int N, const float* __restrict__ vel,
    const float* __restrict__ ew1, const float* __restrict__ eb1,
    const float* __restrict__ ew2, const float* __restrict__ eb2,
    float* __restrict__ h, unsigned short* __restrict__ hbf16)
{
  __shared__ float sp[NPOSC];
  __shared__ float hidl[HID];
  int n = blockIdx.x;
  int j = threadIdx.x;
  if (j < NPOSC){
    const float* v = vel + (size_t)n*(NPOSC*3) + j*3;
    sp[j] = sqrtf(v[0]*v[0] + v[1]*v[1] + v[2]*v[2]);
  }
  __syncthreads();
  float a = eb1[j];
#pragma unroll
  for (int k = 0; k < NPOSC; ++k) a += sp[k] * ew1[k*HID + j];
  hidl[j] = silu(a);
  __syncthreads();
  float o = eb2[j];
  for (int k = 0; k < HID; ++k) o += hidl[k] * ew2[k*HID + j];
  size_t idx = (size_t)n*HID + j;
  h[idx] = o;
  hbf16[idx] = bf16s(o);
}

__global__ void egnn_cov_init(int N, const float* __restrict__ loc,
                              const float* __restrict__ vel, float* __restrict__ cov){
  int i = blockIdx.x * blockDim.x + threadIdx.x;
  int total = N * NCOVC * 3;
  if (i >= total) return;
  int n = i / (NCOVC*3);
  int off = i % (NCOVC*3);
  cov[i] = (off < 30) ? loc[(size_t)n*30 + off] : vel[(size_t)n*30 + (off - 30)];
}

__global__ void egnn_deg(int E, const int* __restrict__ dst, float* __restrict__ deg){
  int i = blockIdx.x * blockDim.x + threadIdx.x;
  if (i >= E) return;
  atomicAdd(&deg[dst[i]], 1.f);
}
__global__ void egnn_invcnt(int N, float* __restrict__ deg){
  int i = blockIdx.x * blockDim.x + threadIdx.x;
  if (i >= N) return;
  deg[i] = 1.f / fmaxf(deg[i], 1.f);
}

// ---------------- edge kernel: msg MLP + pos weights + scatter (all WMMA) ----------------
__global__ __launch_bounds__(128) void egnn_edge(
    const int* __restrict__ srcI, const int* __restrict__ dstI, int E,
    const float* __restrict__ cov,
    const unsigned* __restrict__ hbf,      // h in bf16, read as packed pairs
    const unsigned* __restrict__ w1B,      // msg w1: 4 tiles x 5 chunks
    const unsigned* __restrict__ w2B,      // msg w2: 4 tiles x 2 chunks
    const unsigned* __restrict__ pwB,      // pos_w : 2 tiles x 2 chunks
    const float* __restrict__ b1, const float* __restrict__ b2,
    const float* __restrict__ posb,
    float* __restrict__ m_acc, float* __restrict__ cov_acc)
{
  __shared__ alignas(16) float s_rel [4][16][NCOVC][3];
  __shared__ alignas(16) float s_dist[4][16][32];     // [20..31] zero padding
  __shared__ int   s_src [4][16];
  __shared__ int   s_dst [4][16];
  __shared__ alignas(16) unsigned short s_hid[4][16][HID];

  const int lane = threadIdx.x & 31;
  const int w    = threadIdx.x >> 5;
  const int half = lane >> 4;
  const int m    = lane & 15;
  const long eb  = ((long)blockIdx.x * 4 + w) * 16;
  if (eb >= E) return;

  if (lane < 16){
    long e = eb + lane; if (e >= E) e = E - 1;
    s_src[w][lane] = srcI[e];
    s_dst[w][lane] = dstI[e];
  }
  // zero padding region of s_dist (cols 20..31)
#pragma unroll
  for (int z = lane; z < 16*12; z += 32)
    s_dist[w][z/12][20 + (z%12)] = 0.f;
  wave_sync();

  const int sn = s_src[w][m];
  const int dn = s_dst[w][m];

  // rel + squared-dist: lane handles edge row m, channels half*10..half*10+9
  {
    // 30 consecutive floats per half-row; 8B-aligned -> float2 loads
    const float2* cs2 = (const float2*)(cov + (size_t)sn*(NCOVC*3) + 30*half);
    const float2* cd2 = (const float2*)(cov + (size_t)dn*(NCOVC*3) + 30*half);
    float csv[30], cdv[30];
#pragma unroll
    for (int j = 0; j < 15; ++j){
      float2 a = cs2[j], b = cd2[j];
      csv[2*j] = a.x; csv[2*j+1] = a.y;
      cdv[2*j] = b.x; cdv[2*j+1] = b.y;
    }
#pragma unroll
    for (int c = 0; c < 10; ++c){
      int ch = half*10 + c;
      float r0 = csv[3*c+0] - cdv[3*c+0];
      float r1 = csv[3*c+1] - cdv[3*c+1];
      float r2 = csv[3*c+2] - cdv[3*c+2];
      s_rel[w][m][ch][0] = r0;
      s_rel[w][m][ch][1] = r1;
      s_rel[w][m][ch][2] = r2;
      s_dist[w][m][ch] = r0*r0 + r1*r1 + r2*r2;
    }
  }
  wave_sync();

  // ---- GEMM1: X(16x148) * W1 -> hidden(16x64), K padded to 160 ----
  // A rows are blockwise contiguous in the source rows: 16B segments at uint4
  // index {4c+half, 4c+2+half} of the 128B row.
  const uint4* hs = (const uint4*)(hbf + (((size_t)sn * HID) >> 1));
  const uint4* hd = (const uint4*)(hbf + (((size_t)dn * HID) >> 1));
  v8f acc[4];
#pragma unroll
  for (int t = 0; t < 4; ++t) acc[t] = vzero8();
#pragma unroll
  for (int c = 0; c < 5; ++c){
    B16 A;
    if (c < 4){
      const uint4* hrow = (c < 2) ? hs : hd;
      int cc = c & 1;
      fillA_u(A, hrow[4*cc + half], hrow[4*cc + 2 + half]);
    } else {
      const float4* drow = (const float4*)&s_dist[w][m][0];
      float4 q0 = drow[2*half],     q1 = drow[2*half + 1];
      float4 q2 = drow[4 + 2*half], q3 = drow[4 + 2*half + 1];
      A.u[0]=bf16pk(q0.x,q0.y); A.u[1]=bf16pk(q0.z,q0.w);
      A.u[2]=bf16pk(q1.x,q1.y); A.u[3]=bf16pk(q1.z,q1.w);
      A.u[4]=bf16pk(q2.x,q2.y); A.u[5]=bf16pk(q2.z,q2.w);
      A.u[6]=bf16pk(q3.x,q3.y); A.u[7]=bf16pk(q3.z,q3.w);
    }
#pragma unroll
    for (int t = 0; t < 4; ++t)
      acc[t] = wmma_bf16(A.v, loadB(w1B, t*5 + c, lane), acc[t]);
  }
  // bias + SiLU -> LDS (transpose C layout -> A layout via LDS)
#pragma unroll
  for (int t = 0; t < 4; ++t){
    float bb = b1[t*16 + m];
#pragma unroll
    for (int r = 0; r < 8; ++r)
      s_hid[w][r + half*8][t*16 + m] = bf16s(silu(acc[t][r] + bb));
  }
  wave_sync();

  // ---- GEMM2: hidden(16x64) * W2 -> m(16x64) ----
  const uint4* hrowL = (const uint4*)&s_hid[w][m][0];
  v8f mc[4];
#pragma unroll
  for (int t = 0; t < 4; ++t) mc[t] = vzero8();
#pragma unroll
  for (int c = 0; c < 2; ++c){
    B16 A;
    fillA_u(A, hrowL[4*c + half], hrowL[4*c + 2 + half]);
#pragma unroll
    for (int t = 0; t < 4; ++t)
      mc[t] = wmma_bf16(A.v, loadB(w2B, t*2 + c, lane), mc[t]);
  }

  int  dstRow[8];
  bool valRow[8];
#pragma unroll
  for (int r = 0; r < 8; ++r){
    int row = r + half*8;
    dstRow[r] = s_dst[w][row];
    valRow[r] = (eb + row) < E;
  }
  wave_sync();

  // m: atomic scatter into m_acc; also stash bf16(m) in LDS for pos GEMM
#pragma unroll
  for (int t = 0; t < 4; ++t){
    int ch = t*16 + m;
    float bb = b2[ch];
#pragma unroll
    for (int r = 0; r < 8; ++r){
      float v = mc[t][r] + bb;
      int row = r + half*8;
      if (valRow[r]) atomicAdd(&m_acc[(size_t)dstRow[r]*HID + ch], v);
      s_hid[w][row][ch] = bf16s(v);
    }
  }
  wave_sync();

  // ---- GEMM3: m(16x64) * pos_w(64x20->32) -> w(16x20) ----
  v8f wc[2]; wc[0] = vzero8(); wc[1] = vzero8();
#pragma unroll
  for (int c = 0; c < 2; ++c){
    B16 A;
    fillA_u(A, hrowL[4*c + half], hrowL[4*c + 2 + half]);
#pragma unroll
    for (int t = 0; t < 2; ++t)
      wc[t] = wmma_bf16(A.v, loadB(pwB, t*2 + c, lane), wc[t]);
  }
  // equivariant scatter: cov_acc[dst] += rel * w
#pragma unroll
  for (int t = 0; t < 2; ++t){
    int ch = t*16 + m;
    if (ch < NCOVC){
      float bb = posb[ch];
#pragma unroll
      for (int r = 0; r < 8; ++r){
        if (!valRow[r]) continue;
        int row = r + half*8;
        float wv = wc[t][r] + bb;
        size_t base = (size_t)dstRow[r]*(NCOVC*3) + ch*3;
        atomicAdd(&cov_acc[base+0], s_rel[w][row][ch][0] * wv);
        atomicAdd(&cov_acc[base+1], s_rel[w][row][ch][1] * wv);
        atomicAdd(&cov_acc[base+2], s_rel[w][row][ch][2] * wv);
      }
    }
  }
}

// ---------------- node kernel: h += MLP2([h, m_agg]); cov += cov_acc/cnt ----------------
__global__ __launch_bounds__(32) void egnn_node(
    int N,
    float* __restrict__ h, unsigned short* __restrict__ hbf16,
    float* __restrict__ cov,
    const float* __restrict__ m_acc, const float* __restrict__ cov_acc,
    const float* __restrict__ inv_cnt,
    const unsigned* __restrict__ nw1B,   // 4 tiles x 4 chunks
    const unsigned* __restrict__ nw2B,   // 4 tiles x 2 chunks
    const float* __restrict__ nb1, const float* __restrict__ nb2)
{
  __shared__ alignas(16) unsigned short s_hid[16][HID];
  const int lane = threadIdx.x;
  const int half = lane >> 4;
  const int m    = lane & 15;
  const long nb  = (long)blockIdx.x * 16;
  int nid = (int)(nb + m); if (nid >= N) nid = N - 1;
  const float ic = inv_cnt[nid];
  const unsigned* hb32 = (const unsigned*)hbf16;

  const uint4*  hn   = (const uint4*) (hb32 + (((size_t)nid * HID) >> 1));
  const float4* mrow = (const float4*)(m_acc + (size_t)nid * HID);

  v8f acc[4];
#pragma unroll
  for (int t = 0; t < 4; ++t) acc[t] = vzero8();
#pragma unroll
  for (int c = 0; c < 4; ++c){
    B16 A;
    if (c < 2){
      fillA_u(A, hn[4*c + half], hn[4*c + 2 + half]);
    } else {
      int cc = c - 2;
      float4 p0 = mrow[8*cc + 2*half];
      float4 p1 = mrow[8*cc + 2*half + 1];
      float4 p2 = mrow[8*cc + 4 + 2*half];
      float4 p3 = mrow[8*cc + 4 + 2*half + 1];
      A.u[0]=bf16pk(p0.x*ic,p0.y*ic); A.u[1]=bf16pk(p0.z*ic,p0.w*ic);
      A.u[2]=bf16pk(p1.x*ic,p1.y*ic); A.u[3]=bf16pk(p1.z*ic,p1.w*ic);
      A.u[4]=bf16pk(p2.x*ic,p2.y*ic); A.u[5]=bf16pk(p2.z*ic,p2.w*ic);
      A.u[6]=bf16pk(p3.x*ic,p3.y*ic); A.u[7]=bf16pk(p3.z*ic,p3.w*ic);
    }
#pragma unroll
    for (int t = 0; t < 4; ++t)
      acc[t] = wmma_bf16(A.v, loadB(nw1B, t*4 + c, lane), acc[t]);
  }
#pragma unroll
  for (int t = 0; t < 4; ++t){
    float bb = nb1[t*16 + m];
#pragma unroll
    for (int r = 0; r < 8; ++r)
      s_hid[r + half*8][t*16 + m] = bf16s(silu(acc[t][r] + bb));
  }
  wave_sync();
  const uint4* hrowL = (const uint4*)&s_hid[m][0];
  v8f acc2[4];
#pragma unroll
  for (int t = 0; t < 4; ++t) acc2[t] = vzero8();
#pragma unroll
  for (int c = 0; c < 2; ++c){
    B16 A;
    fillA_u(A, hrowL[4*c + half], hrowL[4*c + 2 + half]);
#pragma unroll
    for (int t = 0; t < 4; ++t)
      acc2[t] = wmma_bf16(A.v, loadB(nw2B, t*2 + c, lane), acc2[t]);
  }
#pragma unroll
  for (int t = 0; t < 4; ++t){
    int ch = t*16 + m;
    float bb = nb2[ch];
#pragma unroll
    for (int r = 0; r < 8; ++r){
      long row = nb + r + half*8;
      if (row < N){
        size_t o = (size_t)row*HID + ch;
        float nh = h[o] + acc2[t][r] + bb;
        h[o] = nh;
        hbf16[o] = bf16s(nh);
      }
    }
  }
  // cov += cov_acc * inv_cnt  (16 nodes x 60)
  for (int i = lane; i < 16*NCOVC*3; i += 32){
    long node = nb + i / (NCOVC*3);
    if (node < N){
      size_t o = (size_t)node*(NCOVC*3) + (i % (NCOVC*3));
      cov[o] += cov_acc[o] * inv_cnt[node];
    }
  }
}

// ---------------- losses ----------------
__global__ __launch_bounds__(256) void egnn_loss(
    int N, int B, int nper,
    const float* __restrict__ cov, const float* __restrict__ loc,
    const float* __restrict__ y, float* __restrict__ out)
{
  __shared__ float sa[256], sf[256];
  int b = blockIdx.x, t = threadIdx.x;
  float a = 0.f, f = 0.f;
  int total = nper * NPOSC;
  for (int i = t; i < total; i += 256){
    int node = b*nper + i / NPOSC;
    int fr   = i % NPOSC;
    size_t pc = (size_t)node*(NCOVC*3) + fr*3;
    size_t lc = (size_t)node*30 + fr*3;
    float dx = cov[pc+0] + loc[lc+0] - y[lc+0];
    float dy = cov[pc+1] + loc[lc+1] - y[lc+1];
    float dz = cov[pc+2] + loc[lc+2] - y[lc+2];
    float d = sqrtf(dx*dx + dy*dy + dz*dz);
    a += d;
    if (fr == NPOSC-1) f += d;
  }
  sa[t] = a; sf[t] = f;
  __syncthreads();
  for (int s = 128; s > 0; s >>= 1){
    if (t < s){ sa[t] += sa[t+s]; sf[t] += sf[t+s]; }
    __syncthreads();
  }
  if (t == 0){
    out[1 + b]     = sa[0] / (float)(nper * NPOSC);
    out[1 + B + b] = sf[0] / (float)nper;
  }
}

__global__ __launch_bounds__(128) void egnn_loss_mean(int B, float* __restrict__ out){
  __shared__ float s[128];
  int t = threadIdx.x;
  float a = 0.f;
  for (int i = t; i < B; i += 128) a += out[1 + i];
  s[t] = a; __syncthreads();
  for (int st = 64; st > 0; st >>= 1){ if (t < st) s[t] += s[t+st]; __syncthreads(); }
  if (t == 0) out[0] = s[0] / (float)B;
}

// =======================================================================
extern "C" void kernel_launch(void* const* d_in, const int* in_sizes, int n_in,
                              void* d_out, int out_size, void* d_ws, size_t ws_size,
                              hipStream_t stream)
{
  (void)n_in; (void)ws_size;
  const int N = in_sizes[0] / 30;
  const int E = in_sizes[3] / 2;
  const int B = (out_size - 1) / 2;
  const int nper = N / B;

  const float* loc = (const float*)d_in[0];
  const float* vel = (const float*)d_in[1];
  const float* y   = (const float*)d_in[2];
  const int*   ei  = (const int*)d_in[3];
  const int*   srcI = ei;
  const int*   dstI = ei + E;
  float* out = (float*)d_out;

  // Param pytree flatten order: sorted-key (jax default) vs insertion order
  const bool sorted = (in_sizes[5] == 64);
  const float *ew1, *eb1, *ew2, *eb2;
  if (sorted){ eb1=(const float*)d_in[5]; eb2=(const float*)d_in[6];
               ew1=(const float*)d_in[7]; ew2=(const float*)d_in[8]; }
  else       { ew1=(const float*)d_in[5]; eb1=(const float*)d_in[6];
               ew2=(const float*)d_in[7]; eb2=(const float*)d_in[8]; }

  struct LP { const float *mw1,*mb1,*mw2,*mb2,*nw1,*nb1,*nw2,*nb2,*pw,*pb; } lp[4];
  for (int i = 0; i < 4; ++i){
    const int bs = 9 + i*10;
    if (sorted){
      lp[i].mb1=(const float*)d_in[bs+0]; lp[i].mb2=(const float*)d_in[bs+1];
      lp[i].mw1=(const float*)d_in[bs+2]; lp[i].mw2=(const float*)d_in[bs+3];
      lp[i].nb1=(const float*)d_in[bs+4]; lp[i].nb2=(const float*)d_in[bs+5];
      lp[i].nw1=(const float*)d_in[bs+6]; lp[i].nw2=(const float*)d_in[bs+7];
      lp[i].pb =(const float*)d_in[bs+8]; lp[i].pw =(const float*)d_in[bs+9];
    } else {
      lp[i].mw1=(const float*)d_in[bs+0]; lp[i].mb1=(const float*)d_in[bs+1];
      lp[i].mw2=(const float*)d_in[bs+2]; lp[i].mb2=(const float*)d_in[bs+3];
      lp[i].nw1=(const float*)d_in[bs+4]; lp[i].nb1=(const float*)d_in[bs+5];
      lp[i].nw2=(const float*)d_in[bs+6]; lp[i].nb2=(const float*)d_in[bs+7];
      lp[i].pw =(const float*)d_in[bs+8]; lp[i].pb =(const float*)d_in[bs+9];
    }
  }

  // ---- workspace carve (256B aligned) ----
  char* p = (char*)d_ws;
  auto take = [&](size_t bytes)->char*{
    char* r = p; p += (bytes + 255) & ~(size_t)255; return r;
  };
  float*          h       = (float*)         take((size_t)N*HID*4);
  unsigned short* hbf     = (unsigned short*)take((size_t)N*HID*2);
  float*          cov     = (float*)         take((size_t)N*NCOVC*3*4);
  float*          m_acc   = (float*)         take((size_t)N*HID*4);
  float*          cov_acc = (float*)         take((size_t)N*NCOVC*3*4);
  float*          inv_cnt = (float*)         take((size_t)N*4);
  // per layer: msgW1(4x5), msgW2(4x2), posW(2x2), nodeW1(4x4), nodeW2(4x2), 256 u32 per block
  const int blkU32[5] = {4*5*256, 4*2*256, 2*2*256, 4*4*256, 4*2*256};
  unsigned* wB[4][5];
  for (int i = 0; i < 4; ++i)
    for (int j = 0; j < 5; ++j)
      wB[i][j] = (unsigned*)take((size_t)blkU32[j]*4);

  // ---- weight pre-swizzle (deterministic, every call) ----
  auto prep = [&](const float* w, unsigned* dstB, int K, int Nw, int nT, int nC){
    int total = nT*nC*256;
    prep_wB<<<(total + 255)/256, 256, 0, stream>>>(w, dstB, K, Nw, nT, nC);
  };
  for (int i = 0; i < 4; ++i){
    prep(lp[i].mw1, wB[i][0], 2*HID + NCOVC, HID,   4, 5);
    prep(lp[i].mw2, wB[i][1], HID,           HID,   4, 2);
    prep(lp[i].pw , wB[i][2], HID,           NCOVC, 2, 2);
    prep(lp[i].nw1, wB[i][3], 2*HID,         HID,   4, 4);
    prep(lp[i].nw2, wB[i][4], HID,           HID,   4, 2);
  }

  // ---- init: embed, cov, degree ----
  egnn_embed<<<N, 64, 0, stream>>>(N, vel, ew1, eb1, ew2, eb2, h, hbf);
  {
    int tot = N*NCOVC*3;
    egnn_cov_init<<<(tot + 255)/256, 256, 0, stream>>>(N, loc, vel, cov);
  }
  hipMemsetAsync(inv_cnt, 0, (size_t)N*4, stream);
  egnn_deg   <<<(E + 255)/256, 256, 0, stream>>>(E, dstI, inv_cnt);
  egnn_invcnt<<<(N + 255)/256, 256, 0, stream>>>(N, inv_cnt);

  // ---- 4 EGNN layers ----
  const int etiles  = (E + 15)/16;
  const int eblocks = (etiles + 3)/4;
  const int ntiles  = (N + 15)/16;
  for (int i = 0; i < 4; ++i){
    hipMemsetAsync(m_acc,   0, (size_t)N*HID*4,     stream);
    hipMemsetAsync(cov_acc, 0, (size_t)N*NCOVC*3*4, stream);
    egnn_edge<<<eblocks, 128, 0, stream>>>(
        srcI, dstI, E, cov, (const unsigned*)hbf,
        wB[i][0], wB[i][1], wB[i][2],
        lp[i].mb1, lp[i].mb2, lp[i].pb, m_acc, cov_acc);
    egnn_node<<<ntiles, 32, 0, stream>>>(
        N, h, hbf, cov, m_acc, cov_acc, inv_cnt,
        wB[i][3], wB[i][4], lp[i].nb1, lp[i].nb2);
  }

  // ---- losses ----
  egnn_loss<<<B, 256, 0, stream>>>(N, B, nper, cov, loc, y, out);
  egnn_loss_mean<<<1, 128, 0, stream>>>(B, out);
}
